// MarkovRandomFieldLoss_37898791419904
// MI455X (gfx1250) — compile-verified, matrix operand order checked by reading, not run on hardware
//
#include <hip/hip_runtime.h>
#include <hip/hip_bf16.h>

// ---------------- problem constants (from reference) ----------------
#define NPATCH 8836      // 94*94 patches per image
#define NPAD   8960      // padded to multiple of 256 (GEMM tiling)
#define KDIM   2304      // 256 channels * 3 * 3
#define GRIDW  94        // patches per row
#define IMGW   96
#define THRESH 0.01f
#define EPSF   1e-8f

#define BCOLS  128       // B slab columns per block
#define BSTR   40        // padded halves per column in LDS (80B stride, bank-conflict free)

typedef __attribute__((ext_vector_type(16))) _Float16 v16h;
typedef __attribute__((ext_vector_type(8)))  _Float16 v8h;
typedef __attribute__((ext_vector_type(8)))  float    v8f;

// Async global->LDS path: inline asm on gfx1250 (ISA: VDST = LDS byte address
// VGPR, VADDR = 64-bit global address), sync copy elsewhere (host parse).
#if defined(__gfx1250__)
#define ASYNC_MODE 1
#else
#define ASYNC_MODE 0
#endif

__device__ __forceinline__ void async_copy_b128(const _Float16* src, _Float16* lds_dst) {
#if ASYNC_MODE == 1
  unsigned ldsa = (unsigned)(uintptr_t)lds_dst;        // low 32 bits = LDS offset
  unsigned long long ga = (unsigned long long)(uintptr_t)src;
  asm volatile("global_load_async_to_lds_b128 %0, %1, off"
               :: "v"(ldsa), "v"(ga) : "memory");
#else
  *(v8h*)lds_dst = *(const v8h*)src;                   // sync fallback
#endif
}

__device__ __forceinline__ void wait_async_le2() {
#if ASYNC_MODE == 1
  asm volatile("s_wait_asynccnt 0x2" ::: "memory");
#endif
}
__device__ __forceinline__ void wait_async_le0() {
#if ASYNC_MODE == 1
  asm volatile("s_wait_asynccnt 0x0" ::: "memory");
#endif
}

// Order-preserving float <-> uint transform (for packed atomic max).
__device__ __forceinline__ unsigned f2mono(float f) {
  unsigned b = __float_as_uint(f);
  return (b & 0x80000000u) ? ~b : (b | 0x80000000u);
}
__device__ __forceinline__ float mono2f(unsigned m) {
  unsigned b = (m & 0x80000000u) ? (m & 0x7fffffffu) : ~m;
  return __uint_as_float(b);
}

// ---------------- kernel 0: zero the packed best[] array ----------------
__global__ void init_best(unsigned long long* __restrict__ best) {
  int i = blockIdx.x * 256 + threadIdx.x;
  if (i < NPAD) best[i] = 0ull;   // any real cosine packs to > 0
}

// ---------------- kernel 1: extract + L2-normalize patches to f16 ----------
__global__ __launch_bounds__(256) void normalize_patches(
    const float* __restrict__ xin, const float* __restrict__ yin,
    _Float16* __restrict__ xn, _Float16* __restrict__ yn) {
  const int p = blockIdx.x;
  const float* img = blockIdx.y ? yin : xin;
  _Float16* dst = blockIdx.y ? yn : xn;
  const int t = threadIdx.x;

  if (p >= NPATCH) {
#pragma unroll
    for (int s = 0; s < 9; ++s)
      dst[(size_t)p * KDIM + t + 256 * s] = (_Float16)0.f;
    return;
  }
  const int r = p / GRIDW, c = p % GRIDW;
  float v[9];
  float ss = 0.f;
#pragma unroll
  for (int s = 0; s < 9; ++s) {
    int e = t + 256 * s;                  // e = ch*9 + dr*3 + dc (unfold order)
    int ch = e / 9, rem = e % 9;
    int dr = rem / 3, dc = rem % 3;
    float f = img[(size_t)ch * (IMGW * IMGW) + (size_t)(r + dr) * IMGW + (c + dc)];
    v[s] = f;
    ss += f * f;
  }
#pragma unroll
  for (int off = 16; off > 0; off >>= 1) ss += __shfl_down(ss, off, 32);
  __shared__ float red[8];
  __shared__ float s_scale;
  if ((t & 31) == 0) red[t >> 5] = ss;
  __syncthreads();
  if (t == 0) {
    float tot = 0.f;
#pragma unroll
    for (int w = 0; w < 8; ++w) tot += red[w];
    s_scale = 1.f / (sqrtf(tot) + EPSF);
  }
  __syncthreads();
  float sc = s_scale;
#pragma unroll
  for (int s = 0; s < 9; ++s)
    dst[(size_t)p * KDIM + t + 256 * s] = (_Float16)(v[s] * sc);
}

// ---------------- kernel 2: WMMA GEMM + fused column max/argmax ----------
// A fragment straight from global; 16-bit A/B layout per ISA 7.12.2:
//   lanes 0-15  : row (lane&15), halves K+0..7  and K+16..23
//   lanes 16-31 : row (lane&15), halves K+8..15 and K+24..31
__device__ __forceinline__ v16h load_fragA(const _Float16* __restrict__ rowbase,
                                           int lane, int k0) {
  const int r = lane & 15;
  const int koff = k0 + ((lane & 16) >> 1);
  const _Float16* p = rowbase + (size_t)r * KDIM + koff;
  union { struct { v8h lo, hi; } h; v16h v; } u;
  u.h.lo = *(const v8h*)(p);
  u.h.hi = *(const v8h*)(p + 16);
  return u.v;
}

// B fragment from the LDS slab: column-major, BSTR-half padded stride.
__device__ __forceinline__ v16h load_fragB(const _Float16* colbase, int lane) {
  const _Float16* p = colbase + ((lane & 16) >> 1);   // +8 halves upper group
  union { struct { v8h lo, hi; } h; v16h v; } u;
  u.h.lo = *(const v8h*)(p);          // ds_load_b128
  u.h.hi = *(const v8h*)(p + 16);     // ds_load_b128
  return u.v;
}

// Stage one 128-col x 32-half B slab (8 KB) into LDS: 512 x 16B transfers,
// 2 per thread, via async global->LDS.
__device__ __forceinline__ void stage_b(_Float16* bs, const _Float16* __restrict__ xn,
                                        int col0, int k0, int td) {
#pragma unroll
  for (int s = 0; s < 2; ++s) {
    int q = td + 256 * s;
    int col = q >> 2;                 // 4 transfers (64B) per column
    int seg = (q & 3) * 8;            // halves within the 32-half chunk
    const _Float16* src = xn + (size_t)(col0 + col) * KDIM + k0 + seg;
    async_copy_b128(src, bs + col * BSTR + seg);
  }
}

__global__ __launch_bounds__(256) void gemm_argmax(
    const _Float16* __restrict__ yn, const _Float16* __restrict__ xn,
    unsigned long long* __restrict__ best) {
  const int td = threadIdx.x;
  const int lane = td & 31;
  const int wave = td >> 5;
  const int rg = wave >> 1;           // row pair 0..3
  const int cg = wave & 1;            // column group 0..1 (64 cols each)
  // Block tile: 128 y-rows (i) x 128 x-cols (j).
  // Wave owns strips (rg*2+s)*16 for s=0,1 and 4 col tiles in its group.
  const int row0 = blockIdx.x * 128 + (rg * 2 + 0) * 16;
  const int row1 = blockIdx.x * 128 + (rg * 2 + 1) * 16;
  const int col0 = blockIdx.y * BCOLS;

  __shared__ __align__(16) _Float16 Bs[2][BCOLS * BSTR];  // 2 x 10 KB

  v8f zero = {0.f, 0.f, 0.f, 0.f, 0.f, 0.f, 0.f, 0.f};
  v8f acc[4][2];
#pragma unroll
  for (int tt = 0; tt < 4; ++tt) { acc[tt][0] = zero; acc[tt][1] = zero; }

  const _Float16* arow0 = yn + (size_t)row0 * KDIM;
  const _Float16* arow1 = yn + (size_t)row1 * KDIM;

  // One k-chunk of compute out of LDS buffer `bb` (holds 2 B frags at a time
  // to minimize interference with the 8 in-place accumulators).
  auto compute_chunk = [&](const _Float16* bb, int k0) {
    v16h a0 = load_fragA(arow0, lane, k0);
    v16h a1 = load_fragA(arow1, lane, k0);
    const _Float16* bbase = bb + (cg * 64 + (lane & 15)) * BSTR;
    {
      v16h b0 = load_fragB(bbase + 0 * 16 * BSTR, lane);
      v16h b1 = load_fragB(bbase + 1 * 16 * BSTR, lane);
      acc[0][0] = __builtin_amdgcn_wmma_f32_16x16x32_f16(false, a0, false, b0, (short)0, acc[0][0], false, false);
      acc[0][1] = __builtin_amdgcn_wmma_f32_16x16x32_f16(false, a1, false, b0, (short)0, acc[0][1], false, false);
      acc[1][0] = __builtin_amdgcn_wmma_f32_16x16x32_f16(false, a0, false, b1, (short)0, acc[1][0], false, false);
      acc[1][1] = __builtin_amdgcn_wmma_f32_16x16x32_f16(false, a1, false, b1, (short)0, acc[1][1], false, false);
    }
    {
      v16h b2 = load_fragB(bbase + 2 * 16 * BSTR, lane);
      v16h b3 = load_fragB(bbase + 3 * 16 * BSTR, lane);
      acc[2][0] = __builtin_amdgcn_wmma_f32_16x16x32_f16(false, a0, false, b2, (short)0, acc[2][0], false, false);
      acc[2][1] = __builtin_amdgcn_wmma_f32_16x16x32_f16(false, a1, false, b2, (short)0, acc[2][1], false, false);
      acc[3][0] = __builtin_amdgcn_wmma_f32_16x16x32_f16(false, a0, false, b3, (short)0, acc[3][0], false, false);
      acc[3][1] = __builtin_amdgcn_wmma_f32_16x16x32_f16(false, a1, false, b3, (short)0, acc[3][1], false, false);
    }
  };

  // prologue: chunk 0 -> buffer 0
  stage_b(Bs[0], xn, col0, 0, td);

  const int NCH = KDIM / 32;          // 72 k-chunks; steady loop stages next
  for (int kc = 0; kc < NCH - 1; ++kc) {
    const int buf = kc & 1;
    stage_b(Bs[buf ^ 1], xn, col0, (kc + 1) * 32, td);
    wait_async_le2();                 // chunk kc complete (in-order), next in flight
    __syncthreads();                  // all waves' slab writes visible
    compute_chunk(Bs[buf], kc * 32);
    __syncthreads();                  // done reading buf before it is restaged
  }
  // peeled final chunk
  wait_async_le0();
  __syncthreads();
  compute_chunk(Bs[(NCH - 1) & 1], (NCH - 1) * 32);

  // Epilogue: C/D layout — lane n (n<16): VGPR v holds (M=v, N=n);
  // lane n+16: (M=v+8, N=n). Reduce over M (y index) per column.
  const int mbase = (lane & 16) ? 8 : 0;
#pragma unroll
  for (int t = 0; t < 4; ++t) {
    unsigned long long cand = 0ull;
#pragma unroll
    for (int s = 0; s < 2; ++s) {
      v8f d = acc[t][s];
      float bv = d[0];
      int bm = 0;
#pragma unroll
      for (int m = 1; m < 8; ++m)
        if (d[m] > bv) { bv = d[m]; bm = m; }   // strict > keeps first index
      int il = bm + mbase;
      float ov = __shfl_xor(bv, 16, 32);
      int oil = __shfl_xor(il, 16, 32);
      if (ov > bv || (ov == bv && oil < il)) { bv = ov; il = oil; }
      int gi = (s ? row1 : row0) + il;
      unsigned long long pv =
          ((unsigned long long)f2mono(bv) << 32) | (unsigned)(~(unsigned)gi);
      if (pv > cand) cand = pv;
    }
    if (lane < 16) {
      // packed compare: bigger value wins; on tie, smaller i (bigger ~i) wins
      atomicMax(&best[col0 + cg * 64 + t * 16 + lane], cand);
    }
  }
}

// ---------------- kernel 3: masked squared distance per x-patch ----------
__global__ __launch_bounds__(256) void patch_loss(
    const float* __restrict__ xin, const float* __restrict__ yin,
    const unsigned long long* __restrict__ best,
    float* __restrict__ partial) {
  const int j = blockIdx.x;
  const int t = threadIdx.x;
  unsigned long long pv = best[j];
  float val = mono2f((unsigned)(pv >> 32));
  int i = (int)(~(unsigned)pv);
  float ss = 0.f;
  if (val >= THRESH && i >= 0 && i < NPATCH) {   // uniform across block
    int rj = j / GRIDW, cj = j % GRIDW;
    int ri = i / GRIDW, ci = i % GRIDW;
#pragma unroll
    for (int s = 0; s < 9; ++s) {
      int e = t + 256 * s;
      int ch = e / 9, rem = e % 9;
      int dr = rem / 3, dc = rem % 3;
      float xv = xin[(size_t)ch * (IMGW * IMGW) + (size_t)(rj + dr) * IMGW + (cj + dc)];
      float yv = yin[(size_t)ch * (IMGW * IMGW) + (size_t)(ri + dr) * IMGW + (ci + dc)];
      float df = xv - yv;
      ss += df * df;
    }
  }
#pragma unroll
  for (int off = 16; off > 0; off >>= 1) ss += __shfl_down(ss, off, 32);
  __shared__ float red[8];
  if ((t & 31) == 0) red[t >> 5] = ss;
  __syncthreads();
  if (t == 0) {
    float tot = 0.f;
#pragma unroll
    for (int w = 0; w < 8; ++w) tot += red[w];
    partial[j] = tot;
  }
}

// ---------------- kernel 4: deterministic final reduction ----------------
__global__ __launch_bounds__(256) void reduce_partials(
    const float* __restrict__ partial, float* __restrict__ out, int n) {
  __shared__ float red[256];
  float s = 0.f;
  for (int t = threadIdx.x; t < n; t += 256) s += partial[t];
  red[threadIdx.x] = s;
  __syncthreads();
  for (int off = 128; off > 0; off >>= 1) {
    if ((int)threadIdx.x < off) red[threadIdx.x] += red[threadIdx.x + off];
    __syncthreads();
  }
  if (threadIdx.x == 0) out[0] = red[0];
}

extern "C" void kernel_launch(void* const* d_in, const int* in_sizes, int n_in,
                              void* d_out, int out_size, void* d_ws, size_t ws_size,
                              hipStream_t stream) {
  const float* x = (const float*)d_in[0];
  const float* y = (const float*)d_in[1];
  char* ws = (char*)d_ws;
  // workspace layout (82.7 MB total):
  _Float16* xn = (_Float16*)ws;                                   // 41.3 MB
  _Float16* yn = (_Float16*)(ws + (size_t)NPAD * KDIM * 2);       // 41.3 MB
  unsigned long long* best =
      (unsigned long long*)(ws + (size_t)NPAD * KDIM * 4);        // 70 KB
  float* partial =
      (float*)(ws + (size_t)NPAD * KDIM * 4 + (size_t)NPAD * 8);  // 35 KB
  float* out = (float*)d_out;

  init_best<<<dim3((NPAD + 255) / 256), dim3(256), 0, stream>>>(best);
  normalize_patches<<<dim3(NPAD, 2), dim3(256), 0, stream>>>(x, y, xn, yn);
  gemm_argmax<<<dim3(NPAD / 128, NPAD / BCOLS), dim3(256), 0, stream>>>(yn, xn, best);
  patch_loss<<<dim3(NPATCH), dim3(256), 0, stream>>>(x, y, best, partial);
  reduce_partials<<<dim3(1), dim3(256), 0, stream>>>(partial, out, NPATCH);
}
